// PoolingFineNet_52948356825727
// MI455X (gfx1250) — matrix operand order repaired and checked by reading, not compiled, not run on hardware
//
#include <hip/hip_runtime.h>

// ---------------- CDNA5 WMMA types ----------------
typedef _Float16 v16h __attribute__((ext_vector_type(16)));
typedef _Float16 v8h  __attribute__((ext_vector_type(8)));
typedef _Float16 v4h  __attribute__((ext_vector_type(4)));
typedef float    v8f  __attribute__((ext_vector_type(8)));
union V16 { v16h v; v8h h[2]; };

// vectorizable-source flags for the gather
#define VF_XA 1
#define VF_XB 2
#define VF_EA 4
#define VF_EB 8

// ---------------- quaternion helpers ----------------
struct Q { float w, x, y, z; };
__device__ __forceinline__ Q qload(const float* p) { return Q{p[0], p[1], p[2], p[3]}; }
__device__ __forceinline__ Q qmul(Q a, Q b) {
  Q r;
  r.w = a.w*b.w - a.x*b.x - a.y*b.y - a.z*b.z;
  r.x = a.w*b.x + a.x*b.w + a.y*b.z - a.z*b.y;
  r.y = a.w*b.y - a.x*b.z + a.y*b.w + a.z*b.x;
  r.z = a.w*b.z + a.x*b.y - a.y*b.x + a.z*b.w;
  return r;
}
__device__ __forceinline__ Q qinv(Q a) { return Q{a.w, -a.x, -a.y, -a.z}; }
__device__ __forceinline__ Q qnorm(Q a) {
  float n = sqrtf(a.w*a.w + a.x*a.x + a.y*a.y + a.z*a.z);
  float r = 1.0f / fmaxf(n, 1e-12f);
  return Q{a.w*r, a.x*r, a.y*r, a.z*r};
}

__device__ __forceinline__ void atomicMaxF(float* addr, float v) {
  int* ai = (int*)addr;
  int cur = *ai;
  while (__int_as_float(cur) < v) {
    int prev = atomicCAS(ai, cur, __float_as_int(v));
    if (prev == cur) break;
    cur = prev;
  }
}

// ---------------- small utility kernels ----------------
__global__ void k_fill(float* p, float v, long n) {
  long i = (long)blockIdx.x * blockDim.x + threadIdx.x;
  if (i < n) p[i] = v;
}
__global__ void k_relu(float* p, long n) {
  long i = (long)blockIdx.x * blockDim.x + threadIdx.x;
  if (i < n) p[i] = fmaxf(p[i], 0.0f);
}
__global__ void k_copy(float* __restrict__ dst, const float* __restrict__ src, long n) {
  long i = (long)blockIdx.x * blockDim.x + threadIdx.x;
  if (i < n) dst[i] = src[i];
}
__global__ void k_upattr(const float* __restrict__ x, const int* __restrict__ gi,
                         const int* __restrict__ gj, const float* __restrict__ ea,
                         int E, float* __restrict__ out) {
  int e = blockIdx.x * blockDim.x + threadIdx.x;
  if (e >= E) return;
  Q xi = qload(x + 4*(size_t)gi[e]);
  Q xj = qload(x + 4*(size_t)gj[e]);
  Q a  = qload(ea + 4*(size_t)e);
  Q r  = qnorm(qmul(qinv(xj), qmul(a, xi)));   // norm(qinv(x[j]) * (ea * x[i]))
  float* o = out + 4*(size_t)e;
  o[0]=r.w; o[1]=r.x; o[2]=r.y; o[3]=r.z;
}
__global__ void k_cnt(const int* __restrict__ gi, const float* __restrict__ emask,
                      int E, float* __restrict__ cnt) {
  int e = blockIdx.x * blockDim.x + threadIdx.x;
  if (e >= E) return;
  atomicAdd(&cnt[gi[e]], emask ? emask[e] : 1.0f);
}
// agg / cnt with relu; writes aligned scratch (primary) and optional d_out mirror
__global__ void k_aggfin(const float* __restrict__ agg, const float* __restrict__ cnt,
                         int Nn, int dout, float* __restrict__ x1, float* __restrict__ x2) {
  long i = (long)blockIdx.x * blockDim.x + threadIdx.x;
  if (i >= (long)Nn * dout) return;
  int n = (int)(i / dout);
  float v = fmaxf(agg[i] / fmaxf(cnt[n], 1.0f), 0.0f);
  x1[i] = v;
  if (x2) x2[i] = v;
}

// ---------------- WMMA edge-conv MLP ----------------
// m = [xA[i], xB[i], xA[j], xB[j], eA[e]*esc, eB[e]*esc]  (din cols, padded to DINP)
// h = relu(m @ W1 + b1) ; e = (h @ W2 + b2) * emask ; agg[i] += e (atomics)
template<int DINP, int DOUT>
__global__ __launch_bounds__(128) void k_econv(
    const float* __restrict__ xA, int dxa,
    const float* __restrict__ xB, int dxb,
    const float* __restrict__ eA, int dea,
    const float* __restrict__ eB, int deb,
    const float* __restrict__ escale,
    const int* __restrict__ gi, const int* __restrict__ gj,
    const float* __restrict__ emask,
    const float* __restrict__ W1, const float* __restrict__ b1,
    const float* __restrict__ W2, const float* __restrict__ b2,
    int din, int E, int vecflags,
    float* __restrict__ e_out, float* __restrict__ agg)
{
  static_assert(DINP % 32 == 0 && DOUT % 32 == 0 && DOUT <= DINP, "shape");
  constexpr int KT1 = DINP / 32;
  constexpr int NT  = DOUT / 16;
  constexpr int KT2 = DOUT / 32;
  constexpr int QD  = DINP / 4;

  __shared__ _Float16 sW1t[DOUT * DINP];   // [n][k] transposed, f16
  __shared__ _Float16 sW2t[DOUT * DOUT];   // [n][k] transposed, f16
  __shared__ _Float16 sM[64 * DINP];       // 64 edge rows, f16 (later aliased by h)
  __shared__ int   sI[64];
  __shared__ int   sJ[64];
  __shared__ float sMask[64];
  __shared__ float sScale[64];
  __shared__ float sB1[DOUT];
  __shared__ float sB2[DOUT];

  const int tid = threadIdx.x;
  const int eBase = blockIdx.x * 64;

  __builtin_prefetch(W1, 0, 1);            // global_prefetch_b8
  __builtin_prefetch(W2, 0, 1);

  for (int r = tid; r < 64; r += 128) {
    int e = eBase + r;
    if (e < E) {
      sI[r] = gi[e]; sJ[r] = gj[e];
      sMask[r]  = emask  ? emask[e]  : 1.0f;
      sScale[r] = escale ? escale[e] : 1.0f;
    } else { sI[r] = 0; sJ[r] = 0; sMask[r] = 0.0f; sScale[r] = 0.0f; }
  }
  for (int c = tid; c < DOUT; c += 128) { sB1[c] = b1[c]; sB2[c] = b2[c]; }
  for (int idx = tid; idx < DOUT * DINP; idx += 128) {
    int n = idx / DINP, k = idx % DINP;
    sW1t[idx] = (_Float16)((k < din) ? W1[(size_t)k * DOUT + n] : 0.0f);
  }
  for (int idx = tid; idx < DOUT * DOUT; idx += 128) {
    int n = idx / DOUT, k = idx % DOUT;
    sW2t[idx] = (_Float16)W2[(size_t)k * DOUT + n];
  }
  __syncthreads();

  // ---- gather m rows in float4 quanta (all segment widths are multiples of 4) ----
  const int dxab = dxa + dxb;
  const int seg2 = dxab + dxa;
  const int seg3 = 2 * dxab;
  const int seg4 = 2 * dxab + dea;
  for (int idx = tid; idx < 64 * QD; idx += 128) {
    int r = idx / QD, k = (idx % QD) * 4;
    int e = eBase + r;
    float v0 = 0.f, v1 = 0.f, v2 = 0.f, v3 = 0.f;
    if (e < E && k < din) {
      int i = sI[r], j = sJ[r];
      const float* src;
      bool vec;
      float sc = 1.0f;
      if (k < dxa)       { src = xA + (size_t)i * dxa + k;              vec = (vecflags & VF_XA); }
      else if (k < dxab) { src = xB + (size_t)i * dxb + (k - dxa);      vec = (vecflags & VF_XB); }
      else if (k < seg2) { src = xA + (size_t)j * dxa + (k - dxab);     vec = (vecflags & VF_XA); }
      else if (k < seg3) { src = xB + (size_t)j * dxb + (k - dxab - dxa); vec = (vecflags & VF_XB); }
      else if (k < seg4) { src = eA + (size_t)e * dea + (k - seg3);     vec = (vecflags & VF_EA); sc = sScale[r]; }
      else               { src = eB + (size_t)e * deb + (k - seg4);     vec = (vecflags & VF_EB); sc = sScale[r]; }
      if (vec) { float4 t = *(const float4*)src; v0 = t.x; v1 = t.y; v2 = t.z; v3 = t.w; }
      else     { v0 = src[0]; v1 = src[1]; v2 = src[2]; v3 = src[3]; }
      v0 *= sc; v1 *= sc; v2 *= sc; v3 *= sc;
    }
    v4h h;
    h.x = (_Float16)v0; h.y = (_Float16)v1; h.z = (_Float16)v2; h.w = (_Float16)v3;
    *(v4h*)(sM + r * DINP + k) = h;                 // ds_store_b64
  }
  __syncthreads();

  const int wave = tid >> 5;
  const int lane = tid & 31;
  const int l15  = lane & 15;
  const int half = lane >> 4;

  const v8f z8 = {0.f,0.f,0.f,0.f,0.f,0.f,0.f,0.f};
  v8f acc[NT];
  #pragma unroll
  for (int nt = 0; nt < NT; ++nt) acc[nt] = z8;

  // ---- first GEMM: m @ W1 ----
  const _Float16* Arow = sM + (wave * 16 + l15) * DINP;
  #pragma unroll
  for (int t = 0; t < KT1; ++t) {
    V16 a;  // A 16x32 f16: elems 0-7 -> K = half*8+.., elems 8-15 -> K = 16+half*8+..
    a.h[0] = *(const v8h*)(Arow + t * 32 + half * 8);
    a.h[1] = *(const v8h*)(Arow + t * 32 + 16 + half * 8);
    #pragma unroll
    for (int nt = 0; nt < NT; ++nt) {
      const _Float16* Brow = sW1t + (l15 + nt * 16) * DINP;
      V16 b;  // B 32x16 f16: elems 0-7 -> K = half*8+.., elems 8-15 -> K = 16+half*8+..
      b.h[0] = *(const v8h*)(Brow + t * 32 + half * 8);
      b.h[1] = *(const v8h*)(Brow + t * 32 + 16 + half * 8);
      acc[nt] = __builtin_amdgcn_wmma_f32_16x16x32_f16(false, a.v, false, b.v,
                                                       (short)0, acc[nt], false, false);
    }
  }

  // ---- bias + relu, stage h (f16) into this wave's (now dead) sM rows ----
  _Float16* H = sM + wave * 16 * DINP;   // 16 x DOUT, stride DOUT
  #pragma unroll
  for (int nt = 0; nt < NT; ++nt) {
    int col = l15 + nt * 16;
    float bb = sB1[col];
    #pragma unroll
    for (int v = 0; v < 8; ++v) {
      float hv = fmaxf(acc[nt][v] + bb, 0.0f);     // C layout: m = half*8+v, n = col
      H[(half * 8 + v) * DOUT + col] = (_Float16)hv;
    }
  }
  __builtin_amdgcn_wave_barrier();   // per-wave LDS ordering; stop compiler reordering

  // ---- second GEMM: h @ W2 ----
  v8f acc2[NT];
  #pragma unroll
  for (int nt = 0; nt < NT; ++nt) acc2[nt] = z8;
  const _Float16* A2 = H + l15 * DOUT;
  #pragma unroll
  for (int t = 0; t < KT2; ++t) {
    V16 a;
    a.h[0] = *(const v8h*)(A2 + t * 32 + half * 8);
    a.h[1] = *(const v8h*)(A2 + t * 32 + 16 + half * 8);
    #pragma unroll
    for (int nt = 0; nt < NT; ++nt) {
      const _Float16* Brow = sW2t + (l15 + nt * 16) * DOUT;
      V16 b;
      b.h[0] = *(const v8h*)(Brow + t * 32 + half * 8);
      b.h[1] = *(const v8h*)(Brow + t * 32 + 16 + half * 8);
      acc2[nt] = __builtin_amdgcn_wmma_f32_16x16x32_f16(false, a.v, false, b.v,
                                                        (short)0, acc2[nt], false, false);
    }
  }

  // ---- e = (acc2 + b2) * emask ; store + atomic segment-sum into agg ----
  #pragma unroll
  for (int v = 0; v < 8; ++v) {
    int row = wave * 16 + half * 8 + v;
    int e = eBase + row;
    if (e < E) {
      float mk = sMask[row];
      int node = sI[row];
      #pragma unroll
      for (int nt = 0; nt < NT; ++nt) {
        int col = l15 + nt * 16;
        float val = (acc2[nt][v] + sB2[col]) * mk;
        if (e_out) e_out[(size_t)e * DOUT + col] = val;
        atomicAdd(&agg[(size_t)node * DOUT + col], val);
      }
    }
  }
}

// ---------------- GAT / pooling kernels ----------------
__global__ void k_gath(const float* __restrict__ x, const float* __restrict__ W,
                       int din, int Nn, float* __restrict__ h) {
  int n = blockIdx.x * blockDim.x + threadIdx.x;
  if (n >= Nn) return;
  float s = 0.f;
  for (int k = 0; k < din; ++k) s += x[(size_t)n * din + k] * W[k];
  h[n] = s;
}
__global__ void k_galpha(const float* __restrict__ h, const int* __restrict__ gi,
                         const int* __restrict__ gj, const float* __restrict__ emask,
                         const float* __restrict__ asrc, const float* __restrict__ adst,
                         int E, float* __restrict__ alpha, float* __restrict__ amax) {
  int e = blockIdx.x * blockDim.x + threadIdx.x;
  if (e >= E) return;
  float a = h[gj[e]] * asrc[0] + h[gi[e]] * adst[0];
  a = (a > 0.f) ? a : 0.2f * a;                         // leaky_relu 0.2
  float mk = emask ? emask[e] : 1.0f;
  a = (mk > 0.f) ? a : -1e9f;
  alpha[e] = a;
  atomicMaxF(&amax[gi[e]], a);
}
__global__ void k_gexden(const float* __restrict__ alpha, const float* __restrict__ amax,
                         const int* __restrict__ gi, const float* __restrict__ emask,
                         int E, float* __restrict__ ex, float* __restrict__ den) {
  int e = blockIdx.x * blockDim.x + threadIdx.x;
  if (e >= E) return;
  float mk = emask ? emask[e] : 1.0f;
  float v = expf(alpha[e] - amax[gi[e]]) * mk;
  ex[e] = v;
  atomicAdd(&den[gi[e]], v);
}
__global__ void k_gscore(const float* __restrict__ ex, const float* __restrict__ den,
                         const int* __restrict__ gi, const int* __restrict__ gj,
                         const float* __restrict__ h, int E, float* __restrict__ score) {
  int e = blockIdx.x * blockDim.x + threadIdx.x;
  if (e >= E) return;
  atomicAdd(&score[gi[e]], ex[e] / fmaxf(den[gi[e]], 1e-16f) * h[gj[e]]);
}
__global__ void k_gbias(float* score, const float* bias, int Nn) {
  int n = blockIdx.x * blockDim.x + threadIdx.x;
  if (n < Nn) score[n] += bias[0];
}
// exact stable top-k via global rank (matches jax.lax.top_k order), LDS-tiled
__global__ void k_rank(const float* __restrict__ score, int Nn, int K,
                       int* __restrict__ perm, float* __restrict__ topv,
                       int* __restrict__ nidx, float* __restrict__ sel) {
  __shared__ float tile[256];
  int n = blockIdx.x * blockDim.x + threadIdx.x;
  float s = (n < Nn) ? score[n] : 0.0f;
  int rank = 0;
  for (int base = 0; base < Nn; base += 256) {
    int m = base + (int)threadIdx.x;
    tile[threadIdx.x] = (m < Nn) ? score[m] : 0.0f;
    __syncthreads();
    int lim = (Nn - base < 256) ? (Nn - base) : 256;
    for (int t = 0; t < lim; ++t) {
      float sm = tile[t];
      int mm = base + t;
      rank += (sm > s) || (sm == s && mm < n);
    }
    __syncthreads();
  }
  if (n >= Nn) return;
  if (rank < K) { perm[rank] = n; topv[rank] = s; nidx[n] = rank; sel[n] = 1.0f; }
  else          { nidx[n] = 0; sel[n] = 0.0f; }
}
__global__ void k_papply(const float* __restrict__ x, int dout,
                         const int* __restrict__ perm, const float* __restrict__ topv,
                         int K, float* __restrict__ xp) {
  long i = (long)blockIdx.x * blockDim.x + threadIdx.x;
  if (i >= (long)K * dout) return;
  int r = (int)(i / dout), c = (int)(i % dout);
  xp[i] = x[(size_t)perm[r] * dout + c] * tanhf(topv[r]);
}
__global__ void k_pedges(const int* __restrict__ gi, const int* __restrict__ gj,
                         const float* __restrict__ emask,
                         const int* __restrict__ nidx, const float* __restrict__ sel,
                         int E, int* __restrict__ eiout, float* __restrict__ ev) {
  int e = blockIdx.x * blockDim.x + threadIdx.x;
  if (e >= E) return;
  int i = gi[e], j = gj[e];
  eiout[e] = nidx[i];
  eiout[E + e] = nidx[j];
  ev[e] = (emask ? emask[e] : 1.0f) * sel[i] * sel[j];
}
__global__ void k_scadd(float* __restrict__ dst, const float* __restrict__ src,
                        const int* __restrict__ perm, int K, int dout) {
  long i = (long)blockIdx.x * blockDim.x + threadIdx.x;
  if (i >= (long)K * dout) return;
  int r = (int)(i / dout), c = (int)(i % dout);
  dst[(size_t)perm[r] * dout + c] += src[i];
}
__global__ void k_linnorm(const float* __restrict__ x5, const float* __restrict__ W,
                          const float* __restrict__ b, const float* __restrict__ xorg,
                          int Nn, float* __restrict__ xq) {
  int n = blockIdx.x * blockDim.x + threadIdx.x;
  if (n >= Nn) return;
  float a0 = b[0], a1 = b[1], a2 = b[2], a3 = b[3];
  for (int k = 0; k < 32; ++k) {
    float xv = x5[(size_t)n * 32 + k];
    a0 += xv * W[k*4+0]; a1 += xv * W[k*4+1];
    a2 += xv * W[k*4+2]; a3 += xv * W[k*4+3];
  }
  Q q = qnorm(qmul(Q{a0,a1,a2,a3}, qload(xorg + 4*(size_t)n)));
  float* o = xq + 4*(size_t)n;
  o[0]=q.w; o[1]=q.x; o[2]=q.y; o[3]=q.z;
}
__global__ void k_losspart(const float* __restrict__ xq, const float* __restrict__ gt,
                           const int* __restrict__ gi, const int* __restrict__ gj,
                           const float* __restrict__ beta, int E, float* __restrict__ part) {
  __shared__ float red[256];
  int e = blockIdx.x * blockDim.x + threadIdx.x;
  float s = 0.0f;
  if (e < E) {
    int i = gi[e], j = gj[e];
    Q rg = qmul(qinv(qload(gt + 4*(size_t)i)), qload(gt + 4*(size_t)j));
    Q rx = qmul(qinv(qload(xq + 4*(size_t)i)), qload(xq + 4*(size_t)j));
    Q l = qnorm(qmul(qinv(rg), rx));
    float bb = beta[0];
    float dd[4] = {fabsf(l.w - 1.0f), fabsf(l.x), fabsf(l.y), fabsf(l.z)};
    for (int c = 0; c < 4; ++c)
      s += (dd[c] < bb) ? (0.5f * dd[c] * dd[c] / bb) : (dd[c] - 0.5f * bb);
  }
  red[threadIdx.x] = s;
  __syncthreads();
  for (int off = 128; off > 0; off >>= 1) {
    if (threadIdx.x < off) red[threadIdx.x] += red[threadIdx.x + off];
    __syncthreads();
  }
  if (threadIdx.x == 0) part[blockIdx.x] = red[0];
}
__global__ void k_lossfin(const float* __restrict__ part, int nb, int E, float* __restrict__ out) {
  if (threadIdx.x == 0 && blockIdx.x == 0) {
    float s = 0.0f;
    for (int i = 0; i < nb; ++i) s += part[i];
    out[0] = s / (4.0f * (float)E);
  }
}

// ---------------- host side ----------------
static inline int GD(long n) { return (int)((n + 255) / 256); }

template<int DINP, int DOUT>
static void run_conv(const float* xA, int dxa, const float* xB, int dxb,
                     const float* eA, int dea, const float* eB, int deb,
                     const float* escale,
                     const int* gi, const int* gj, const float* emask,
                     const float* W1, const float* b1, const float* W2, const float* b2,
                     int din, int E, int Nn, int vecflags,
                     float* e_out, float* x_out, float* x_mirror,
                     float* agg, float* cnt, hipStream_t s)
{
  long an = (long)Nn * DOUT;
  k_fill<<<GD(an), 256, 0, s>>>(agg, 0.0f, an);
  k_fill<<<GD(Nn), 256, 0, s>>>(cnt, 0.0f, Nn);
  k_cnt<<<GD(E), 256, 0, s>>>(gi, emask, E, cnt);
  k_econv<DINP, DOUT><<<(E + 63) / 64, 128, 0, s>>>(xA, dxa, xB, dxb, eA, dea, eB, deb,
                                                    escale, gi, gj, emask, W1, b1, W2, b2,
                                                    din, E, vecflags, e_out, agg);
  k_aggfin<<<GD(an), 256, 0, s>>>(agg, cnt, Nn, DOUT, x_out, x_mirror);
  if (e_out) k_relu<<<GD((long)E * DOUT), 256, 0, s>>>(e_out, (long)E * DOUT);
}

static void run_gat(const float* x, int din, const int* gi, const int* gj,
                    const float* emask, int Nn, int E,
                    const float* W, const float* asrc, const float* adst, const float* bias,
                    float* hsc, float* alpha, float* exb, float* amax, float* den,
                    float* score, hipStream_t s)
{
  k_fill<<<GD(Nn), 256, 0, s>>>(amax, -1e30f, Nn);
  k_fill<<<GD(Nn), 256, 0, s>>>(den, 0.0f, Nn);
  k_fill<<<GD(Nn), 256, 0, s>>>(score, 0.0f, Nn);
  k_gath<<<GD(Nn), 256, 0, s>>>(x, W, din, Nn, hsc);
  k_galpha<<<GD(E), 256, 0, s>>>(hsc, gi, gj, emask, asrc, adst, E, alpha, amax);
  k_gexden<<<GD(E), 256, 0, s>>>(alpha, amax, gi, emask, E, exb, den);
  k_gscore<<<GD(E), 256, 0, s>>>(exb, den, gi, gj, hsc, E, score);
  k_gbias<<<GD(Nn), 256, 0, s>>>(score, bias, Nn);
}

extern "C" void kernel_launch(void* const* d_in, const int* in_sizes, int n_in,
                              void* d_out, int out_size, void* d_ws, size_t ws_size,
                              hipStream_t stream)
{
  (void)in_sizes; (void)n_in; (void)out_size; (void)ws_size;
  const int N = 10000, E = 320000;
  const int K1 = 5000, K2 = 2500;
  hipStream_t s = stream;

  const float* x_org = (const float*)d_in[0];
  const int*   eix   = (const int*)d_in[1];
  const float* eattr = (const float*)d_in[2];
  const float* gt_q  = (const float*)d_in[3];
  const float* beta  = (const float*)d_in[4];
  auto P = [&](int i) { return (const float*)d_in[i]; };
  // flattened params: conv1=5 conv2=9 conv3=13 sub_pre=17 sub1=21 sub2=25 subsub=29
  // conv4=33 ; pool1 W/as/ad/bias=37..40 ; pool2=41..44 ; lin1 W=45 b=46

  const int* GI = eix;
  const int* GJ = eix + E;

  float* out = (float*)d_out;
  float* OX  = out;                       // (N,4)
  float* OL  = out + (size_t)N * 4;       // loss scalar
  float* OX1 = OL + 1;
  float* OX2 = OX1 + (size_t)N * 32;
  float* OX3 = OX2 + (size_t)N * 32;
  float* OX4 = OX3 + (size_t)N * 32;
  float* OX5 = OX4 + (size_t)N * 32;
  float* OE1 = OX5 + (size_t)N * 32;
  float* OE2 = OE1 + (size_t)E * 32;
  float* OE3 = OE2 + (size_t)E * 32;
  float* OE4 = OE3 + (size_t)E * 32;
  float* OE5 = OE4 + (size_t)E * 32;

  // workspace bump allocator (256B aligned blocks)
  char* wsb = (char*)d_ws;
  size_t off = 0;
  auto A  = [&](size_t elems) -> void* {
    off = (off + 255) & ~(size_t)255;
    void* p = wsb + off;
    off += elems * 4;
    return p;
  };
  auto AF = [&](size_t n) { return (float*)A(n); };
  auto AI = [&](size_t n) { return (int*)A(n); };

  float* eam  = AF((size_t)E * 4);
  float* esA  = AF((size_t)E * 64);   // es1, later reused as ess1
  float* esB  = AF((size_t)E * 64);   // es2
  float* agg  = AF((size_t)N * 64);
  float* cnt  = AF(N);
  float* sx1  = AF((size_t)N * 32);   // aligned mirrors of x1..x4 (gather sources)
  float* sx2  = AF((size_t)N * 32);
  float* sx3  = AF((size_t)N * 32);
  float* sx4  = AF((size_t)N * 32);
  float* xs1  = AF((size_t)N * 64);
  float* xp1  = AF((size_t)K1 * 64);
  float* xs2  = AF((size_t)K1 * 64);
  float* xp2  = AF((size_t)K2 * 64);
  float* xss1 = AF((size_t)K2 * 64);
  float* xss2 = AF((size_t)K2 * 64);
  float* xs3  = AF((size_t)K1 * 32);
  float* hsc  = AF(N);
  float* amax = AF(N);
  float* den  = AF(N);
  float* scr  = AF(N);
  float* alp  = AF(E);
  float* exb  = AF(E);
  float* tv1  = AF(K1);
  float* tv2  = AF(K2);
  float* sel1 = AF(N);
  float* sel2 = AF(K1);
  float* ev1  = AF(E);
  float* ev2  = AF(E);
  float* part = AF(2048);
  int* perm1 = AI(K1);
  int* perm2 = AI(K2);
  int* nidx1 = AI(N);
  int* nidx2 = AI(K1);
  int* ei1   = AI((size_t)2 * E);
  int* ei2   = AI((size_t)2 * E);

  // 1. eam = update_attr(x_org, ei, edge_attr)
  k_upattr<<<GD(E), 256, 0, s>>>(x_org, GI, GJ, eattr, E, eam);

  // 2-4. conv1..conv3 (full graph); x outputs -> aligned scratch + d_out mirror
  run_conv<32, 32>(x_org, 4, nullptr, 0, eam, 4, nullptr, 0, nullptr,
                   GI, GJ, nullptr, P(5), P(6), P(7), P(8), 12, E, N,
                   VF_XA | VF_EA, OE1, sx1, OX1, agg, cnt, s);
  run_conv<128, 32>(sx1, 32, nullptr, 0, eam, 4, OE1, 32, nullptr,
                    GI, GJ, nullptr, P(9), P(10), P(11), P(12), 100, E, N,
                    VF_XA | VF_EA, OE2, sx2, OX2, agg, cnt, s);
  run_conv<192, 32>(sx2, 32, sx1, 32, OE2, 32, OE1, 32, nullptr,
                    GI, GJ, nullptr, P(13), P(14), P(15), P(16), 192, E, N,
                    VF_XA | VF_XB, OE3, sx3, OX3, agg, cnt, s);

  // 5. conv3_sub_pre
  run_conv<96, 64>(sx3, 32, nullptr, 0, OE3, 32, nullptr, 0, nullptr,
                   GI, GJ, nullptr, P(17), P(18), P(19), P(20), 96, E, N,
                   VF_XA, esA, xs1, nullptr, agg, cnt, s);

  // 6. SAG pool 1
  run_gat(xs1, 64, GI, GJ, nullptr, N, E, P(37), P(38), P(39), P(40),
          hsc, alp, exb, amax, den, scr, s);
  k_rank<<<GD(N), 256, 0, s>>>(scr, N, K1, perm1, tv1, nidx1, sel1);
  k_papply<<<GD((long)K1 * 64), 256, 0, s>>>(xs1, 64, perm1, tv1, K1, xp1);
  k_pedges<<<GD(E), 256, 0, s>>>(GI, GJ, nullptr, nidx1, sel1, E, ei1, ev1);

  // 7. conv3_sub1 (ea1 = es1*ev1 via escale)
  run_conv<192, 64>(xp1, 64, nullptr, 0, esA, 64, nullptr, 0, ev1,
                    ei1, ei1 + E, ev1, P(21), P(22), P(23), P(24), 192, E, K1,
                    VF_XA | VF_EA, esB, xs2, nullptr, agg, cnt, s);

  // 8. SAG pool 2
  run_gat(xs2, 64, ei1, ei1 + E, ev1, K1, E, P(41), P(42), P(43), P(44),
          hsc, alp, exb, amax, den, scr, s);
  k_rank<<<GD(K1), 256, 0, s>>>(scr, K1, K2, perm2, tv2, nidx2, sel2);
  k_papply<<<GD((long)K2 * 64), 256, 0, s>>>(xs2, 64, perm2, tv2, K2, xp2);
  k_pedges<<<GD(E), 256, 0, s>>>(ei1, ei1 + E, ev1, nidx2, sel2, E, ei2, ev2);

  // 9. conv3_subsub #1 (ea2 = es2*ev2; ess1 overwrites esA, es1 is dead)
  run_conv<192, 64>(xp2, 64, nullptr, 0, esB, 64, nullptr, 0, ev2,
                    ei2, ei2 + E, ev2, P(29), P(30), P(31), P(32), 192, E, K2,
                    VF_XA | VF_EA, esA, xss1, nullptr, agg, cnt, s);
  // 10. conv3_subsub #2 (e output discarded)
  run_conv<192, 64>(xss1, 64, nullptr, 0, esA, 64, nullptr, 0, nullptr,
                    ei2, ei2 + E, ev2, P(29), P(30), P(31), P(32), 192, E, K2,
                    VF_XA | VF_EA, nullptr, xss2, nullptr, agg, cnt, s);
  // 11. xs2r = xs2.at[perm2].add(xss2)
  k_scadd<<<GD((long)K2 * 64), 256, 0, s>>>(xs2, xss2, perm2, K2, 64);
  // 12. conv3_sub2
  run_conv<192, 32>(xs2, 64, nullptr, 0, esB, 64, nullptr, 0, nullptr,
                    ei1, ei1 + E, ev1, P(25), P(26), P(27), P(28), 192, E, K1,
                    VF_XA | VF_EA, nullptr, xs3, nullptr, agg, cnt, s);
  // 13. x3 = x3.at[perm1].add(xs3); refresh d_out mirror
  k_scadd<<<GD((long)K1 * 32), 256, 0, s>>>(sx3, xs3, perm1, K1, 32);
  k_copy<<<GD((long)N * 32), 256, 0, s>>>(OX3, sx3, (long)N * 32);

  // 14-15. conv4 twice (shared params)
  run_conv<192, 32>(sx3, 32, sx2, 32, OE3, 32, OE2, 32, nullptr,
                    GI, GJ, nullptr, P(33), P(34), P(35), P(36), 192, E, N,
                    VF_XA | VF_XB, OE4, sx4, OX4, agg, cnt, s);
  run_conv<192, 32>(sx4, 32, sx3, 32, OE4, 32, OE3, 32, nullptr,
                    GI, GJ, nullptr, P(33), P(34), P(35), P(36), 192, E, N,
                    VF_XA | VF_XB, OE5, OX5, nullptr, agg, cnt, s);

  // 16. x = norm(qmul(x5 @ lin.W + b, x_org))
  k_linnorm<<<GD(N), 256, 0, s>>>(OX5, P(45), P(46), x_org, N, OX);

  // 17. huber loss (deterministic two-stage reduction)
  int nb = GD(E);
  k_losspart<<<nb, 256, 0, s>>>(OX, gt_q, GI, GJ, beta, E, part);
  k_lossfin<<<1, 1, 0, s>>>(part, nb, E, OL);
}
